// NPUSafeCrossAttnEncoder2d_85950885528384
// MI455X (gfx1250) — matrix operand order repaired and checked by reading, not compiled, not run on hardware
//
#include <hip/hip_runtime.h>
#include <hip/hip_bf16.h>

// ---------------------------------------------------------------------------
// Fused cross-attention encoder for MI455X (gfx1250, wave32, WMMA bf16 + TDM).
//
// Algebraic folds (exact): k_b dropped (softmax shift-invariant);
//   QKT = k_w^T q_w, kc = q_b^T k_w  -> scores = (seed^T QKT + kc) emb
//   OV  = out_w v_w, ob = out_w v_b + out_b -> hidden_pre = OV (emb w^T) + ob
// RMSNorm fold: rms depends only on the GEMM N dim -> scale D by rms[f] in the
//   epilogue, so x is read from HBM exactly once (~210 MB).
// emb tile restage for the pooled GEMM runs on the Tensor Data Mover
//   (tensor_load_to_lds, 2-D descriptor with LDS row padding), overlapped with
//   the softmax, fenced by s_wait_tensorcnt.
// Workspace requirement: ~224 MB.
// ---------------------------------------------------------------------------

#define DI __device__ __forceinline__

typedef __bf16 bf16;
typedef __attribute__((ext_vector_type(8)))  bf16  bf16x8;
typedef __attribute__((ext_vector_type(16))) bf16  v16bf;
typedef __attribute__((ext_vector_type(8)))  float v8f;
typedef __attribute__((ext_vector_type(4)))  unsigned int u32x4;
typedef __attribute__((ext_vector_type(8)))  int  i32x8;
typedef __attribute__((ext_vector_type(4)))  int  i32x4;

#if __has_include(<hip/amd_detail/amd_gfx1250_TDM.h>)
#define TDM_SIX_ARGS 1
#endif

constexpr int   Bn = 2, Cn = 256, Tn = 400, Fn = 256, Kn = 32;
constexpr int   TF = Tn * Fn;
constexpr float EPSV = 1e-6f;
constexpr int   PAD = 264;   // padded row stride (elements) for 256-wide LDS tiles
constexpr int   HSTR = 520;  // padded row stride for the [32][512] FFN hidden tile

DI int lane_id() { return threadIdx.x & 31; }
DI int wave_id() { return threadIdx.x >> 5; }

// ---- WMMA fragment helpers (layouts per CDNA5 ISA 7.12.2) ------------------
DI v16bf frag_A(const bf16* base, int stride, int row0, int kb) {
  int l = lane_id();
  const bf16* p = base + (size_t)(row0 + (l & 15)) * stride + kb + ((l >> 4) << 3);
  bf16x8 lo = *(const bf16x8*)p;
  bf16x8 hi = *(const bf16x8*)(p + 16);
  return __builtin_shufflevector(lo, hi, 0,1,2,3,4,5,6,7,8,9,10,11,12,13,14,15);
}
DI v16bf frag_BT(const bf16* base, int stride, int row0, int kb) {
  int l = lane_id();
  const bf16* p = base + (size_t)(row0 + (l & 15)) * stride + kb + ((l >> 4) << 3);
  bf16x8 lo = *(const bf16x8*)p;
  bf16x8 hi = *(const bf16x8*)(p + 16);
  return __builtin_shufflevector(lo, hi, 0,1,2,3,4,5,6,7,8,9,10,11,12,13,14,15);
}
DI v8f wmma_bf(v16bf a, v16bf b, v8f c) {
  return __builtin_amdgcn_wmma_f32_16x16x32_bf16(false, a, false, b, (short)0, c, false, false);
}
DI void storeD_T_bf16(bf16* base, int stride, int m0, int n0, v8f acc) {
  int l = lane_id();
  int n = n0 + (l & 15), m = m0 + ((l >> 4) << 3);
  bf16x8 t;
#pragma unroll
  for (int r = 0; r < 8; ++r) t[r] = (bf16)acc[r];
  *(bf16x8*)(base + (size_t)n * stride + m) = t;
}
DI void storeD_T_f32_bias(float* base, int stride, int m0, int n0, v8f acc,
                          const float* bias) {
  int l = lane_id();
  int n = n0 + (l & 15), m = m0 + ((l >> 4) << 3);
  float* p = base + (size_t)n * stride + m;
#pragma unroll
  for (int r = 0; r < 8; ++r) p[r] = acc[r] + bias[m + r];
}
DI float silu(float x) { return x * (1.f / (1.f + __expf(-x))); }

// ---- Tensor Data Mover: async 2-D tile load global->LDS --------------------
// Loads tile_d1 rows of tile_d0 bf16 elements (row stride stride0 elements in
// global memory) into LDS at lds_off, inserting pad_amt DWORDs of LDS padding
// after every pad_int_code'd interval (6 -> 128 DWORDs = one 256-elem row).
DI void tdm_load_2d_bf16(unsigned lds_off, const void* gptr,
                         unsigned tile_d0, unsigned tile_d1,
                         unsigned tensor_d0, unsigned tensor_d1,
                         unsigned stride0) {
  unsigned long long ga = (unsigned long long)(size_t)gptr;
  u32x4 g0;
  g0[0] = 1u;                                   // count=1 (valid user D#)
  g0[1] = lds_off;                              // LDS byte address
  g0[2] = (unsigned)ga;                         // global addr [31:0]
  g0[3] = (unsigned)((ga >> 32) & 0x1FFFFFFu)   // global addr [56:32]
          | (2u << 30);                         // type = image
  i32x8 g1;
  g1[0] = (int)((1u << 16)    // data_size = 1 -> 2 bytes
              | (1u << 20)    // pad_enable
              | (6u << 22)    // pad_interval: 128 DWORDs (512 B row)
              | (3u << 25));  // pad_amount: 4 DWORDs (16 B -> PAD=264)
  g1[1] = (int)((tensor_d0 & 0xFFFFu) << 16);                     // dim0 lo16
  g1[2] = (int)(((tensor_d0 >> 16) & 0xFFFFu) |
                ((tensor_d1 & 0xFFFFu) << 16));                   // dim0 hi/dim1 lo
  g1[3] = (int)(((tensor_d1 >> 16) & 0xFFFFu) |
                ((tile_d0 & 0xFFFFu) << 16));                     // dim1 hi/tile0
  g1[4] = (int)(tile_d1 & 0xFFFFu);                               // tile1, tile2=0
  g1[5] = (int)stride0;                                           // dim0 stride lo32
  g1[6] = 0;
  g1[7] = 0;
  i32x4 z4 = {0, 0, 0, 0};
#if defined(TDM_SIX_ARGS)
  i32x8 z8 = {0, 0, 0, 0, 0, 0, 0, 0};
  __builtin_amdgcn_tensor_load_to_lds(g0, g1, z4, z4, z8, 0);
#else
  __builtin_amdgcn_tensor_load_to_lds(g0, g1, z4, z4, 0);
#endif
}

// ---------------------------------------------------------------------------
// Prep kernels
// ---------------------------------------------------------------------------
__global__ void k_f32_to_bf16(const float* __restrict__ src, bf16* __restrict__ dst, int n) {
  for (int i = blockIdx.x * blockDim.x + threadIdx.x; i < n; i += gridDim.x * blockDim.x)
    dst[i] = (bf16)src[i];
}
__global__ void k_prep_qkov(const float* __restrict__ q_w, const float* __restrict__ k_w,
                            const float* __restrict__ out_w, const float* __restrict__ v_w,
                            bf16* __restrict__ QKT, bf16* __restrict__ OV) {
  int tid = threadIdx.x;
  if (blockIdx.x < Cn) {
    int i = blockIdx.x;
    float s = 0.f;
    for (int o = 0; o < Cn; ++o) s += k_w[o * Cn + i] * q_w[o * Cn + tid];
    QKT[i * Cn + tid] = (bf16)s;
  } else {
    int c = blockIdx.x - Cn;
    float s = 0.f;
    for (int o = 0; o < Cn; ++o) s += out_w[c * Cn + o] * v_w[o * Cn + tid];
    OV[c * Cn + tid] = (bf16)s;
  }
}
__global__ void k_prep_vec(const float* __restrict__ q_b, const float* __restrict__ k_w,
                           const float* __restrict__ out_w, const float* __restrict__ v_b,
                           const float* __restrict__ out_b,
                           float* __restrict__ kc, float* __restrict__ ob) {
  int i = threadIdx.x;
  float s0 = 0.f, s1 = 0.f;
  for (int o = 0; o < Cn; ++o) {
    s0 += q_b[o] * k_w[o * Cn + i];
    s1 += out_w[i * Cn + o] * v_b[o];
  }
  kc[i] = s0;
  ob[i] = s1 + out_b[i];
}

// ---------------------------------------------------------------------------
// K1: RMSNorm(channel) + 1x1 conv GEMM, one workgroup per (b,t) frame.
// Single pass over x: stage x*w[c] (bf16) transposed, accumulate sumsq, and
// apply rms[f] to the GEMM result columns in the epilogue.
// ---------------------------------------------------------------------------
constexpr size_t SMEM1 = (size_t)Fn * PAD * 2 + (size_t)Fn * 4;

__global__ void __launch_bounds__(256)
k_rmsnorm_pw(const float* __restrict__ x, const float* __restrict__ norm_w,
             const bf16* __restrict__ Wpre, const float* __restrict__ pw_b,
             bf16* __restrict__ emb0) {
  extern __shared__ char sm[];
  bf16*  xnT  = (bf16*)sm;                          // [Fn][PAD]
  float* rmsL = (float*)(sm + (size_t)Fn * PAD * 2);  // [Fn]

  int bid = blockIdx.x;
  int b = bid / Tn, t = bid - b * Tn;
  int tid = threadIdx.x;  // tid == f
  size_t base = (size_t)b * Cn * TF + (size_t)t * Fn;

  float ss = 0.f;
  for (int c = 0; c < Cn; ++c) {
    float v = x[base + (size_t)c * TF + tid];
    ss += v * v;
    xnT[(size_t)tid * PAD + c] = (bf16)(v * norm_w[c]);
  }
  rmsL[tid] = rsqrtf(ss * (1.f / Cn) + EPSV);
  __syncthreads();

  int w = wave_id(), l = lane_id();
  for (int p = 0; p < 4; ++p) {            // f-block passes
    v8f acc[8];
#pragma unroll
    for (int j = 0; j < 8; ++j) acc[j] = v8f{};
    for (int ks = 0; ks < 8; ++ks) {
      __builtin_prefetch(Wpre + (size_t)(2 * w) * 16 * Cn + ((ks + 1) & 7) * 32, 0, 1);
      v16bf a0 = frag_A(Wpre, Cn, (2 * w) * 16, ks * 32);
      v16bf a1 = frag_A(Wpre, Cn, (2 * w + 1) * 16, ks * 32);
#pragma unroll
      for (int j = 0; j < 4; ++j) {
        v16bf bt = frag_BT(xnT, PAD, (p * 4 + j) * 16, ks * 32);
        acc[j]     = wmma_bf(a0, bt, acc[j]);
        acc[4 + j] = wmma_bf(a1, bt, acc[4 + j]);
      }
    }
    int nloc = l & 15, mo = (l >> 4) << 3;
#pragma unroll
    for (int oi = 0; oi < 2; ++oi)
#pragma unroll
      for (int j = 0; j < 4; ++j) {
        int m0 = (2 * w + oi) * 16 + mo;
        int f0 = (p * 4 + j) * 16 + nloc;
        float rms = rmsL[f0];
#pragma unroll
        for (int r = 0; r < 8; ++r) {
          int o = m0 + r;
          emb0[base + (size_t)o * TF + f0] = (bf16)(acc[oi * 4 + j][r] * rms + pw_b[o]);
        }
      }
  }
}

// ---------------------------------------------------------------------------
// K2: causal depthwise 3x3 + bias + SiLU -> emb (f32 out + bf16 ws copy),
// fused with basis pooling: seedT[b][t][k][c] = sum_f basis[k,f]*emb[c,f].
// ---------------------------------------------------------------------------
__global__ void __launch_bounds__(256)
k_dwconv_pool(const bf16* __restrict__ emb0, const float* __restrict__ dw_w,
              const float* __restrict__ dw_b, const float* __restrict__ basis,
              float* __restrict__ emb_out, bf16* __restrict__ emb_bf,
              bf16* __restrict__ seedT) {
  __shared__ float row_s[Fn];
  __shared__ float red2[Kn * 8];

  int bid = blockIdx.x;
  int b = bid / Tn, t = bid - b * Tn;
  int tid = threadIdx.x;  // tid == f
  int kk = tid & 31, part = tid >> 5;

  float breg[32];
#pragma unroll
  for (int j = 0; j < 32; ++j) breg[j] = basis[kk * Fn + part * 32 + j];

  size_t base = (size_t)b * Cn * TF + (size_t)t * Fn;
  size_t seed_base = (size_t)(b * Tn + t) * Kn * Cn;

  for (int c = 0; c < Cn; ++c) {
    const float* w9 = dw_w + c * 9;
    size_t cb = base + (size_t)c * TF;
    float acc = dw_b[c];
#pragma unroll
    for (int kt = 0; kt < 3; ++kt) {
      int tt = t - 2 + kt;
      if (tt < 0) continue;
      const bf16* rowp = emb0 + cb + (long)(tt - t) * Fn;
#pragma unroll
      for (int kf = 0; kf < 3; ++kf) {
        int ff = tid - 1 + kf;
        if (ff >= 0 && ff < Fn) acc += w9[kt * 3 + kf] * (float)rowp[ff];
      }
    }
    float s = silu(acc);
    emb_out[cb + tid] = s;
    emb_bf[cb + tid] = (bf16)s;
    row_s[tid] = s;
    __syncthreads();
    float pp = 0.f;
#pragma unroll
    for (int j = 0; j < 32; ++j) pp += breg[j] * row_s[part * 32 + j];
    red2[kk * 8 + part] = pp;
    __syncthreads();
    if (tid < Kn) {
      float sm2 = 0.f;
#pragma unroll
      for (int p2 = 0; p2 < 8; ++p2) sm2 += red2[tid * 8 + p2];
      seedT[seed_base + (size_t)tid * Cn + c] = (bf16)sm2;
    }
    __syncthreads();
  }
}

// ---------------------------------------------------------------------------
// K4: fused attention + FFN residual. One workgroup (8 waves) per (b,t).
// ---------------------------------------------------------------------------
constexpr size_t OFF_EMB = 0;                                // emb tile [256][PAD] bf16 / hT [32][HSTR] f32
constexpr size_t OFF_QK  = OFF_EMB + (size_t)Cn * PAD * 2;   // qk  [32][PAD] bf16
constexpr size_t OFF_SC  = OFF_QK + (size_t)Kn * PAD * 2;    // scores [32][PAD] f32
constexpr size_t OFF_W   = OFF_SC + (size_t)Kn * PAD * 4;    // softmax w [32][PAD] bf16
constexpr size_t OFF_PT  = OFF_W + (size_t)Kn * PAD * 2;     // pooled^T [32][PAD] bf16
constexpr size_t OFF_HP  = OFF_PT + (size_t)Kn * PAD * 2;    // hidden_pre^T [32][PAD] f32
constexpr size_t OFF_HN  = OFF_HP + (size_t)Kn * PAD * 4;    // rmsnormed^T [32][PAD] bf16
constexpr size_t OFF_G   = OFF_HN + (size_t)Kn * PAD * 2;    // swiglu^T [32][PAD] bf16
constexpr size_t OFF_RED = OFF_G + (size_t)Kn * PAD * 2;     // reductions
constexpr size_t SMEM4   = OFF_RED + Kn * 8 * 4 + Kn * 4;

__global__ void __launch_bounds__(256)
k_attn_ffn(const bf16* __restrict__ emb_bf, const bf16* __restrict__ seedT,
           const bf16* __restrict__ QKT, const float* __restrict__ kc,
           const bf16* __restrict__ OV, const float* __restrict__ ob,
           const float* __restrict__ ffn_norm_w,
           const bf16* __restrict__ FIN, const float* __restrict__ fin_b,
           const bf16* __restrict__ FOUT, const float* __restrict__ fout_b,
           const float* __restrict__ rb, const float* __restrict__ sscale_p,
           const float* __restrict__ pscale_p, float* __restrict__ hidden_out) {
  extern __shared__ char sm[];
  bf16*  embB = (bf16*)(sm + OFF_EMB);
  float* hT   = (float*)(sm + OFF_EMB);
  bf16*  qkL  = (bf16*)(sm + OFF_QK);
  float* scF  = (float*)(sm + OFF_SC);
  bf16*  wsm  = (bf16*)(sm + OFF_W);
  bf16*  pT   = (bf16*)(sm + OFF_PT);
  float* hpT  = (float*)(sm + OFF_HP);
  bf16*  hnT  = (bf16*)(sm + OFF_HN);
  bf16*  gT   = (bf16*)(sm + OFF_G);
  float* red  = (float*)(sm + OFF_RED);
  float* redS = (float*)(sm + OFF_RED + Kn * 8 * 4);

  int bid = blockIdx.x;
  int b = bid / Tn, t = bid - b * Tn;
  int tid = threadIdx.x, l = tid & 31, w = tid >> 5;
  float sscale = *sscale_p, pscale = *pscale_p;
  size_t ebase = (size_t)b * Cn * TF + (size_t)t * Fn;
  const bf16* seed_bt = seedT + (size_t)(b * Tn + t) * Kn * Cn;

  // ---- stage emb^T [f][c] for the scores GEMM (vectorized 16B loads)
  for (int s = 0; s < 32; ++s) {
    int c = s * 8 + (tid >> 5);
    int f0 = (tid & 31) * 8;
    bf16x8 v = *(const bf16x8*)(emb_bf + ebase + (size_t)c * TF + f0);
#pragma unroll
    for (int j = 0; j < 8; ++j) embB[(size_t)(f0 + j) * PAD + c] = v[j];
  }

  // ---- qk GEMM: qk[k,i] = sum_c seed^T[k,c] QKT[i,c]  (A, B^T both global)
  {
    int mb = w & 1, nb0 = (w >> 1) * 4;
    v8f acc[4];
#pragma unroll
    for (int j = 0; j < 4; ++j) acc[j] = v8f{};
    for (int ks = 0; ks < 8; ++ks) {
      v16bf a = frag_A(seed_bt, Cn, mb * 16, ks * 32);
#pragma unroll
      for (int j = 0; j < 4; ++j)
        acc[j] = wmma_bf(a, frag_BT(QKT, Cn, (nb0 + j) * 16, ks * 32), acc[j]);
    }
    int nloc = l & 15, mo = (l >> 4) << 3;
#pragma unroll
    for (int j = 0; j < 4; ++j) {
      int n = (nb0 + j) * 16 + nloc;
      float kcv = kc[n];
#pragma unroll
      for (int r = 0; r < 8; ++r)
        qkL[(mb * 16 + mo + r) * PAD + n] = (bf16)(acc[j][r] + kcv);
    }
  }
  __syncthreads();

  // ---- scores GEMM: scores[k,f] = sum_i qk[k,i] emb[i,f]
  {
    int mb = w & 1, nb0 = (w >> 1) * 4;
    v8f acc[4];
#pragma unroll
    for (int j = 0; j < 4; ++j) acc[j] = v8f{};
    for (int ks = 0; ks < 8; ++ks) {
      v16bf a = frag_A(qkL, PAD, mb * 16, ks * 32);
#pragma unroll
      for (int j = 0; j < 4; ++j)
        acc[j] = wmma_bf(a, frag_BT(embB, PAD, (nb0 + j) * 16, ks * 32), acc[j]);
    }
    int nloc = l & 15, mo = (l >> 4) << 3;
#pragma unroll
    for (int j = 0; j < 4; ++j) {
      int n = (nb0 + j) * 16 + nloc;
#pragma unroll
      for (int r = 0; r < 8; ++r) {
        int m = mb * 16 + mo + r;
        scF[m * PAD + n] = acc[j][r] * sscale + rb[m * Fn + n] * pscale;
      }
    }
  }
  __syncthreads();

  // ---- kick off async TDM reload of emb as [c][f] into the same LDS buffer
  // (scores GEMM is done with emb^T; the DMA overlaps the softmax below).
  if (tid < 32) {
    tdm_load_2d_bf16((unsigned)(size_t)(void*)embB, (const void*)(emb_bf + ebase),
                     /*tile_d0=*/Fn, /*tile_d1=*/Cn,
                     /*tensor_d0=*/Fn, /*tensor_d1=*/Cn, /*stride0=*/TF);
  }

  // ---- softmax over f (rows of scF) -> wsm bf16
  {
    int kk = tid >> 3, seg = tid & 7;
    float mx = -3.4e38f;
#pragma unroll
    for (int j = 0; j < 32; ++j) mx = fmaxf(mx, scF[kk * PAD + seg * 32 + j]);
    red[kk * 8 + seg] = mx;
    __syncthreads();
    if (tid < Kn) {
      float m2 = red[tid * 8];
#pragma unroll
      for (int p = 1; p < 8; ++p) m2 = fmaxf(m2, red[tid * 8 + p]);
      redS[tid] = m2;
    }
    __syncthreads();
    mx = redS[kk];
    float s = 0.f;
#pragma unroll
    for (int j = 0; j < 32; ++j) {
      float e = __expf(scF[kk * PAD + seg * 32 + j] - mx);
      scF[kk * PAD + seg * 32 + j] = e;
      s += e;
    }
    __syncthreads();
    red[kk * 8 + seg] = s;
    __syncthreads();
    if (tid < Kn) {
      float s2 = 0.f;
#pragma unroll
      for (int p = 0; p < 8; ++p) s2 += red[tid * 8 + p];
      redS[tid] = 1.f / s2;
    }
    __syncthreads();
    float inv = redS[kk];
#pragma unroll
    for (int j = 0; j < 32; ++j)
      wsm[kk * PAD + seg * 32 + j] = (bf16)(scF[kk * PAD + seg * 32 + j] * inv);
  }

  // ---- fence the TDM transfer, then barrier before consuming embB=[c][f]
  __builtin_amdgcn_s_wait_tensorcnt(0);
  __syncthreads();

  // ---- pooled GEMM: pooled[c,k] = sum_f emb[c,f] w[k,f] -> pooled^T in LDS
  {
    v8f acc[4];
#pragma unroll
    for (int j = 0; j < 4; ++j) acc[j] = v8f{};
    for (int ks = 0; ks < 8; ++ks) {
      v16bf bt0 = frag_BT(wsm, PAD, 0, ks * 32);
      v16bf bt1 = frag_BT(wsm, PAD, 16, ks * 32);
      v16bf a0 = frag_A(embB, PAD, (2 * w) * 16, ks * 32);
      v16bf a1 = frag_A(embB, PAD, (2 * w + 1) * 16, ks * 32);
      acc[0] = wmma_bf(a0, bt0, acc[0]);
      acc[1] = wmma_bf(a1, bt0, acc[1]);
      acc[2] = wmma_bf(a0, bt1, acc[2]);
      acc[3] = wmma_bf(a1, bt1, acc[3]);
    }
#pragma unroll
    for (int j = 0; j < 4; ++j)
      storeD_T_bf16(pT, PAD, ((2 * w) | (j & 1)) * 16, (j >> 1) * 16, acc[j]);
  }
  __syncthreads();

  // ---- OV GEMM: hidden_pre[c,k] = sum_i OV[c,i] pooled[i,k] + ob[c]
  {
    v8f acc[4];
#pragma unroll
    for (int j = 0; j < 4; ++j) acc[j] = v8f{};
    for (int ks = 0; ks < 8; ++ks) {
      __builtin_prefetch(OV + (size_t)(2 * w) * 16 * Cn + ((ks + 1) & 7) * 32, 0, 1);
      v16bf bt0 = frag_BT(pT, PAD, 0, ks * 32);
      v16bf bt1 = frag_BT(pT, PAD, 16, ks * 32);
      v16bf a0 = frag_A(OV, Cn, (2 * w) * 16, ks * 32);
      v16bf a1 = frag_A(OV, Cn, (2 * w + 1) * 16, ks * 32);
      acc[0] = wmma_bf(a0, bt0, acc[0]);
      acc[1] = wmma_bf(a1, bt0, acc[1]);
      acc[2] = wmma_bf(a0, bt1, acc[2]);
      acc[3] = wmma_bf(a1, bt1, acc[3]);
    }
#pragma unroll
    for (int j = 0; j < 4; ++j)
      storeD_T_f32_bias(hpT, PAD, ((2 * w) | (j & 1)) * 16, (j >> 1) * 16, acc[j], ob);
  }
  __syncthreads();

  // ---- RMSNorm over c of hidden_pre (per k column) -> hnT bf16
  {
    int kk = tid >> 3, seg = tid & 7;
    float ss = 0.f;
#pragma unroll
    for (int j = 0; j < 32; ++j) {
      float v = hpT[kk * PAD + seg * 32 + j];
      ss += v * v;
    }
    red[kk * 8 + seg] = ss;
    __syncthreads();
    if (tid < Kn) {
      float s2 = 0.f;
#pragma unroll
      for (int p = 0; p < 8; ++p) s2 += red[tid * 8 + p];
      redS[tid] = rsqrtf(s2 * (1.f / Cn) + EPSV);
    }
    __syncthreads();
    float rr = redS[kk];
#pragma unroll
    for (int j = 0; j < 32; ++j) {
      int c = seg * 32 + j;
      hnT[kk * PAD + c] = (bf16)(hpT[kk * PAD + c] * rr * ffn_norm_w[c]);
    }
  }
  __syncthreads();

  // ---- FFN GEMM1: h[512,32] = ffn_in_w @ hn (+ b) -> hT f32 (reuses embB LDS)
  {
    v8f acc[8];
#pragma unroll
    for (int j = 0; j < 8; ++j) acc[j] = v8f{};
    for (int ks = 0; ks < 8; ++ks) {
      v16bf bt0 = frag_BT(hnT, PAD, 0, ks * 32);
      v16bf bt1 = frag_BT(hnT, PAD, 16, ks * 32);
#pragma unroll
      for (int jm = 0; jm < 4; ++jm) {
        v16bf a = frag_A(FIN, Cn, (w * 4 + jm) * 16, ks * 32);
        acc[jm]     = wmma_bf(a, bt0, acc[jm]);
        acc[4 + jm] = wmma_bf(a, bt1, acc[4 + jm]);
      }
    }
#pragma unroll
    for (int jm = 0; jm < 4; ++jm) {
      storeD_T_f32_bias(hT, HSTR, (w * 4 + jm) * 16, 0, acc[jm], fin_b);
      storeD_T_f32_bias(hT, HSTR, (w * 4 + jm) * 16, 16, acc[4 + jm], fin_b);
    }
  }
  __syncthreads();

  // ---- SwiGLU: g[i,k] = h[i,k] * silu(h[i+256,k]) -> gT bf16
  {
    int kk = tid >> 3, seg = tid & 7;
#pragma unroll
    for (int j = 0; j < 32; ++j) {
      int i = seg * 32 + j;
      float av = hT[kk * HSTR + i];
      float bv = hT[kk * HSTR + Cn + i];
      gT[kk * PAD + i] = (bf16)(av * silu(bv));
    }
  }
  __syncthreads();

  // ---- FFN GEMM2 + residual -> hidden_out [B,C,T,K]
  {
    v8f acc[4];
#pragma unroll
    for (int j = 0; j < 4; ++j) acc[j] = v8f{};
    for (int ks = 0; ks < 8; ++ks) {
      v16bf bt0 = frag_BT(gT, PAD, 0, ks * 32);
      v16bf bt1 = frag_BT(gT, PAD, 16, ks * 32);
      v16bf a0 = frag_A(FOUT, Cn, (2 * w) * 16, ks * 32);
      v16bf a1 = frag_A(FOUT, Cn, (2 * w + 1) * 16, ks * 32);
      acc[0] = wmma_bf(a0, bt0, acc[0]);
      acc[1] = wmma_bf(a1, bt0, acc[1]);
      acc[2] = wmma_bf(a0, bt1, acc[2]);
      acc[3] = wmma_bf(a1, bt1, acc[3]);
    }
    int nloc = l & 15, mo = (l >> 4) << 3;
#pragma unroll
    for (int j = 0; j < 4; ++j) {
      int m0 = (((2 * w) | (j & 1)) * 16) + mo;
      int n = (j >> 1) * 16 + nloc;
#pragma unroll
      for (int r = 0; r < 8; ++r) {
        int c = m0 + r;
        float val = acc[j][r] + hpT[n * PAD + c] + fout_b[c];
        hidden_out[(((size_t)b * Cn + c) * Tn + t) * Kn + n] = val;
      }
    }
  }
}

// ---------------------------------------------------------------------------
// Host launcher
// ---------------------------------------------------------------------------
extern "C" void kernel_launch(void* const* d_in, const int* in_sizes, int n_in,
                              void* d_out, int out_size, void* d_ws, size_t ws_size,
                              hipStream_t stream) {
  (void)in_sizes; (void)n_in; (void)out_size; (void)ws_size;
  const float* x          = (const float*)d_in[0];
  const float* pre_norm_w = (const float*)d_in[1];
  const float* pre_pw_w   = (const float*)d_in[2];
  const float* pre_pw_b   = (const float*)d_in[3];
  const float* pre_dw_w   = (const float*)d_in[4];
  const float* pre_dw_b   = (const float*)d_in[5];
  const float* q_w        = (const float*)d_in[6];
  const float* q_b        = (const float*)d_in[7];
  const float* k_w        = (const float*)d_in[8];
  /* k_b (d_in[9]) dropped: softmax shift-invariant */
  const float* v_w        = (const float*)d_in[10];
  const float* v_b        = (const float*)d_in[11];
  const float* out_w      = (const float*)d_in[12];
  const float* out_b      = (const float*)d_in[13];
  const float* ffn_norm_w = (const float*)d_in[14];
  const float* ffn_in_w   = (const float*)d_in[15];
  const float* ffn_in_b   = (const float*)d_in[16];
  const float* ffn_out_w  = (const float*)d_in[17];
  const float* ffn_out_b  = (const float*)d_in[18];
  const float* score_scale= (const float*)d_in[19];
  const float* prior_scale= (const float*)d_in[20];
  const float* basis      = (const float*)d_in[21];
  const float* routing_b  = (const float*)d_in[22];

  float* hidden_out = (float*)d_out;
  float* emb_out    = hidden_out + (size_t)Bn * Cn * Tn * Kn;

  // workspace layout (bytes); requires ~224 MB
  char* ws = (char*)d_ws;
  size_t off = 0;
  bf16* emb0   = (bf16*)(ws + off); off += (size_t)Bn * Cn * TF * 2;       // 104.9 MB
  bf16* embbf  = (bf16*)(ws + off); off += (size_t)Bn * Cn * TF * 2;       // 104.9 MB
  bf16* seedT  = (bf16*)(ws + off); off += (size_t)Bn * Tn * Kn * Cn * 2;  // 13.1 MB
  bf16* Wpre   = (bf16*)(ws + off); off += (size_t)Cn * Cn * 2;
  bf16* QKT    = (bf16*)(ws + off); off += (size_t)Cn * Cn * 2;
  bf16* OV     = (bf16*)(ws + off); off += (size_t)Cn * Cn * 2;
  bf16* FIN    = (bf16*)(ws + off); off += (size_t)2 * Cn * Cn * 2;
  bf16* FOUT   = (bf16*)(ws + off); off += (size_t)Cn * Cn * 2;
  float* kc    = (float*)(ws + off); off += Cn * 4;
  float* ob    = (float*)(ws + off); off += Cn * 4;

  hipFuncSetAttribute((const void*)k_rmsnorm_pw,
                      hipFuncAttributeMaxDynamicSharedMemorySize, (int)SMEM1);
  hipFuncSetAttribute((const void*)k_attn_ffn,
                      hipFuncAttributeMaxDynamicSharedMemorySize, (int)SMEM4);

  k_f32_to_bf16<<<128, 256, 0, stream>>>(pre_pw_w, Wpre, Cn * Cn);
  k_f32_to_bf16<<<256, 256, 0, stream>>>(ffn_in_w, FIN, 2 * Cn * Cn);
  k_f32_to_bf16<<<256, 256, 0, stream>>>(ffn_out_w, FOUT, Cn * Cn);
  k_prep_qkov<<<2 * Cn, 256, 0, stream>>>(q_w, k_w, out_w, v_w, QKT, OV);
  k_prep_vec<<<1, 256, 0, stream>>>(q_b, k_w, out_w, v_b, out_b, kc, ob);

  k_rmsnorm_pw<<<Bn * Tn, 256, SMEM1, stream>>>(x, pre_norm_w, Wpre, pre_pw_b, emb0);
  k_dwconv_pool<<<Bn * Tn, 256, 0, stream>>>(emb0, pre_dw_w, pre_dw_b, basis,
                                             emb_out, embbf, seedT);
  k_attn_ffn<<<Bn * Tn, 256, SMEM4, stream>>>(embbf, seedT, QKT, kc, OV, ob,
                                              ffn_norm_w, FIN, ffn_in_b, FOUT,
                                              ffn_out_b, routing_b, score_scale,
                                              prior_scale, hidden_out);
}
